// UniSkip_75282186764695
// MI455X (gfx1250) — matrix operand order repaired and checked by reading, not compiled, not run on hardware
//
#include <hip/hip_runtime.h>

// ---------------------------------------------------------------------------
// GRU-last for B=64, T=48, V=20000, E=620, H=2400 on gfx1250 (MI455X).
// bf16 WMMA (v_wmma_f32_16x16x32_bf16) with fp32 accumulate.
// Recurrent GEMM stages its B operand through LDS with the gfx1250
// async global->LDS path (ASYNCcnt).
// ---------------------------------------------------------------------------

typedef __bf16 bf16;
typedef __attribute__((ext_vector_type(16))) __bf16 v16bf;
typedef __attribute__((ext_vector_type(8)))  __bf16 v8bf;
typedef __attribute__((ext_vector_type(4)))  __bf16 v4bf;
typedef __attribute__((ext_vector_type(8)))  float  v8f;
typedef __attribute__((ext_vector_type(2)))  int    v2i;

// address-space-qualified element types for the async-to-LDS builtin
// (clang prints AS1 as `__device__` and AS3 as `__shared__` in HIP mode)
typedef __attribute__((address_space(1))) v2i as1_v2i;
typedef __attribute__((address_space(3))) v2i as3_v2i;

#define Bsz 64
#define Tsz 48
#define Esz 620
#define EP  640          // E padded to multiple of 32 (WMMA K)
#define Hsz 2400         // multiple of 32 already
#define Gsz 7200         // 3*H
#define ROWS (Bsz * Tsz) // 3072

// recurrent-GEMM LDS staging parameters
#define KS   96                    // k-slab: 3 WMMA k-steps; 2400/96 = 25 slabs
#define NSLAB (Hsz / KS)           // 25
#define LDSB_STRIDE 104            // padded row stride (elems): 52 dwords/row
#define CHUNKS (48 * (KS / 4))     // 48 rows x 24 8-byte pieces = 1152
#define CH_PER_THREAD (CHUNKS / 128) // 9 async ops / thread / slab (exact)

#if __has_builtin(__builtin_amdgcn_s_wait_asynccnt)
#define WAIT_ASYNC(n) __builtin_amdgcn_s_wait_asynccnt(n)
#else
#define WAIT_ASYNC(n) asm volatile("s_wait_asynccnt %0" ::"i"(n) : "memory")
#endif

// ---- WMMA helpers ----------------------------------------------------------

__device__ __forceinline__ v8f wmma_bf16(v16bf a, v16bf b, v8f c) {
  // 8 args: (neg_a, A, neg_b, B, c_mod, C, reuse_a, reuse_b)
  return __builtin_amdgcn_wmma_f32_16x16x32_bf16(false, a, false, b,
                                                 (short)0, c, false, false);
}

// A-matrix 16x32 bf16 fragment (row-major source, stride lda elements).
// Lane L<16 : row m0+L, K = k0+0..15 (elems 0..15)
// Lane L>=16: row m0+L-16, K = k0+16..31
__device__ __forceinline__ v16bf load_a_frag(const bf16* __restrict__ A,
                                             int lda, int m0, int k0, int lane) {
  const int l15 = lane & 15;
  const int hi  = (lane >> 4) & 1;
  const bf16* p = A + (size_t)(m0 + l15) * lda + (k0 + hi * 16);
  union { v16bf v; v8bf h[2]; } u;
  u.h[0] = *(const v8bf*)(p);
  u.h[1] = *(const v8bf*)(p + 8);
  return u.v;
}

// B-matrix 32x16 bf16 fragment from W stored [N x K] row-major (K contiguous):
// B[k,n] = W[n,k]. Lane n<16: column n0+n, K = k0+0..15; lane n+16: K=k0+16..31.
__device__ __forceinline__ v16bf load_b_frag(const bf16* __restrict__ W,
                                             int ldw, int n0, int k0, int lane) {
  const int l15 = lane & 15;
  const int hi  = (lane >> 4) & 1;
  const bf16* p = W + (size_t)(n0 + l15) * ldw + (k0 + hi * 16);
  return *(const v16bf*)p;
}

// Same B fragment but from an LDS-staged slab [48 x LDSB_STRIDE] (16B loads,
// the padded stride is only 16B aligned, not 32B).
__device__ __forceinline__ v16bf load_b_frag_lds(const bf16* Lb, int nf, int kk,
                                                 int lane) {
  const int l15 = lane & 15;
  const int hi  = (lane >> 4) & 1;
  const bf16* p = Lb + (nf + l15) * LDSB_STRIDE + (kk + hi * 16);
  union { v16bf v; v8bf h[2]; } u;
  u.h[0] = *(const v8bf*)(p);       // ds_load_b128
  u.h[1] = *(const v8bf*)(p + 8);   // ds_load_b128
  return u.v;
}

// C/D 16x16 f32: element e of v8f in lane L -> row m0 + e + (L>=16 ? 8 : 0),
// col n0 + (L&15).
__device__ __forceinline__ void store_c(float* __restrict__ C, int ldc,
                                        int m0, int n0, int lane, v8f c) {
  const int col   = n0 + (lane & 15);
  const int rbase = m0 + ((lane >> 4) & 1) * 8;
  float* p = C + (size_t)rbase * ldc + col;
#pragma unroll
  for (int e = 0; e < 8; ++e) p[(size_t)e * ldc] = c[e];
}

// ---- one-time prep kernels -------------------------------------------------

__global__ __launch_bounds__(256) void init_h_kernel(float* __restrict__ Hf,
                                                     bf16* __restrict__ Hb) {
  int idx = blockIdx.x * 256 + threadIdx.x;
  if (idx >= Bsz * Hsz) return;
  Hf[idx] = 0.0f;
  Hb[idx] = (bf16)0.0f;
}

// gather + convert embeddings into padded bf16 [ROWS x EP]
__global__ __launch_bounds__(256) void embed_kernel(const int* __restrict__ tokens,
                                                    const float* __restrict__ emb,
                                                    bf16* __restrict__ Xb) {
  int idx = blockIdx.x * 256 + threadIdx.x;
  if (idx >= ROWS * EP) return;
  int row = idx / EP, k = idx % EP;
  float v = 0.0f;
  if (k < Esz) {
    int tok = tokens[row];                       // in [1, V]; row 0 is padding
    v = emb[(size_t)tok * Esz + k];
  }
  Xb[idx] = (bf16)v;
}

__global__ __launch_bounds__(256) void conv_wih_kernel(const float* __restrict__ W,
                                                       bf16* __restrict__ Wb) {
  int idx = blockIdx.x * 256 + threadIdx.x;
  if (idx >= Gsz * EP) return;
  int n = idx / EP, k = idx % EP;
  Wb[idx] = (bf16)(k < Esz ? W[(size_t)n * Esz + k] : 0.0f);
}

__global__ __launch_bounds__(256) void conv_whh_kernel(const float* __restrict__ W,
                                                       bf16* __restrict__ Wb) {
  int idx = blockIdx.x * 256 + threadIdx.x;
  if (idx >= Gsz * Hsz) return;
  Wb[idx] = (bf16)W[idx];
}

// ---- gi = X @ w_ih^T  :  [3072 x 640] * [7200 x 640]^T -> [3072 x 7200] ----
// 256 threads = 8 waves; wave grid 4(M) x 2(N); wave tile 32x48 (2 A x 3 B).
// Block tile 128 x 96.  grid = (7200/96, 3072/128) = (75, 24).
__global__ __launch_bounds__(256) void gi_gemm_kernel(const bf16* __restrict__ X,
                                                      const bf16* __restrict__ W,
                                                      float* __restrict__ Gi) {
  const int lane = threadIdx.x & 31;
  const int wave = threadIdx.x >> 5;
  const int wm = wave & 3;
  const int wn = wave >> 2;
  const int m0 = blockIdx.y * 128 + wm * 32;
  const int n0 = blockIdx.x * 96  + wn * 48;

  v8f acc[2][3] = {};
  for (int k0 = 0; k0 < EP; k0 += 32) {
    v16bf a0 = load_a_frag(X, EP, m0,      k0, lane);
    v16bf a1 = load_a_frag(X, EP, m0 + 16, k0, lane);
    v16bf b0 = load_b_frag(W, EP, n0,      k0, lane);
    v16bf b1 = load_b_frag(W, EP, n0 + 16, k0, lane);
    v16bf b2 = load_b_frag(W, EP, n0 + 32, k0, lane);
    acc[0][0] = wmma_bf16(a0, b0, acc[0][0]);
    acc[0][1] = wmma_bf16(a0, b1, acc[0][1]);
    acc[0][2] = wmma_bf16(a0, b2, acc[0][2]);
    acc[1][0] = wmma_bf16(a1, b0, acc[1][0]);
    acc[1][1] = wmma_bf16(a1, b1, acc[1][1]);
    acc[1][2] = wmma_bf16(a1, b2, acc[1][2]);
  }
#pragma unroll
  for (int i = 0; i < 2; ++i)
#pragma unroll
    for (int j = 0; j < 3; ++j)
      store_c(Gi, Gsz, m0 + 16 * i, n0 + 16 * j, lane, acc[i][j]);
}

// ---- gh = h @ w_hh^T : [64 x 2400] * [7200 x 2400]^T -> [64 x 7200] --------
// 128 threads = 4 waves; wave w owns rows 16w..16w+15, block owns 48 columns.
// grid = 7200/48 = 150.  w_hh_bf (34.5 MB) is L2-resident across all 48 steps.
// The shared B slab (48 cols x 96 k) is staged into LDS once per block with
// async global->LDS copies, double-buffered against the WMMA work, removing
// the 4x intra-block B-fragment redundancy on the sequential critical path.
__global__ __launch_bounds__(128) void gh_gemm_kernel(const bf16* __restrict__ Hb,
                                                      const bf16* __restrict__ W,
                                                      float* __restrict__ Gh) {
  __shared__ bf16 lds_b[2][48 * LDSB_STRIDE];   // 2 x 9984 B

  const int tid  = threadIdx.x;
  const int lane = tid & 31;
  const int wave = tid >> 5;   // 0..3
  const int m0 = wave * 16;
  const int n0 = blockIdx.x * 48;

  auto issue_copy = [&](int buf, int k0) {
#pragma unroll
    for (int c = 0; c < CH_PER_THREAD; ++c) {
      const int chunk = tid + c * 128;          // 0..1151
      const int n = chunk / (KS / 4);           // row within slab (0..47)
      const int p = chunk % (KS / 4);           // 8-byte piece within row
      const bf16* g = W + (size_t)(n0 + n) * Hsz + k0 + p * 4;
      bf16* l = &lds_b[buf][n * LDSB_STRIDE + p * 4];
#if __has_builtin(__builtin_amdgcn_global_load_async_to_lds_b64)
      __builtin_amdgcn_global_load_async_to_lds_b64((as1_v2i*)g, (as3_v2i*)l,
                                                    0, 0);
#else
      *(v4bf*)l = *(const v4bf*)g;              // global_load_b64 + ds_store_b64
#endif
    }
  };

  v8f acc[3] = {};
  issue_copy(0, 0);
  for (int s = 0; s < NSLAB; ++s) {
    if (s + 1 < NSLAB) {
      issue_copy((s + 1) & 1, (s + 1) * KS);
      WAIT_ASYNC(CH_PER_THREAD);   // slab s complete; slab s+1 may be in flight
    } else {
      WAIT_ASYNC(0);
    }
    __syncthreads();
    const bf16* lb = lds_b[s & 1];
#pragma unroll
    for (int kk = 0; kk < KS; kk += 32) {
      v16bf a  = load_a_frag(Hb, Hsz, m0, s * KS + kk, lane);
      v16bf b0 = load_b_frag_lds(lb, 0,  kk, lane);
      v16bf b1 = load_b_frag_lds(lb, 16, kk, lane);
      v16bf b2 = load_b_frag_lds(lb, 32, kk, lane);
      acc[0] = wmma_bf16(a, b0, acc[0]);
      acc[1] = wmma_bf16(a, b1, acc[1]);
      acc[2] = wmma_bf16(a, b2, acc[2]);
    }
    __syncthreads();  // everyone done reading buf (s&1) before it is refilled
  }
#pragma unroll
  for (int j = 0; j < 3; ++j) store_c(Gh, Gsz, m0, n0 + 16 * j, lane, acc[j]);
}

// ---- fused gate update + last-state scatter --------------------------------
__global__ __launch_bounds__(256) void gate_kernel(const float* __restrict__ Gi,
                                                   const float* __restrict__ Gh,
                                                   const float* __restrict__ bhh,
                                                   const int* __restrict__ lengths,
                                                   float* __restrict__ Hf,
                                                   bf16* __restrict__ Hb,
                                                   float* __restrict__ Out,
                                                   int t) {
  int idx = blockIdx.x * 256 + threadIdx.x;
  if (idx >= Bsz * Hsz) return;
  int b = idx / Hsz, j = idx % Hsz;

  const float* gi = Gi + (size_t)(b * Tsz + t) * Gsz;
  const float* gh = Gh + (size_t)b * Gsz;

  float ghr = gh[j]           + bhh[j];
  float ghz = gh[Hsz + j]     + bhh[Hsz + j];
  float ghn = gh[2 * Hsz + j] + bhh[2 * Hsz + j];

  float r = 1.0f / (1.0f + __expf(-(gi[j] + ghr)));
  float z = 1.0f / (1.0f + __expf(-(gi[Hsz + j] + ghz)));
  float n = tanhf(gi[2 * Hsz + j] + r * ghn);

  float h  = Hf[idx];
  float hn = (1.0f - z) * n + z * h;
  Hf[idx] = hn;
  Hb[idx] = (bf16)hn;                 // A operand for next step's WMMA
  if (t == lengths[b] - 1) Out[idx] = hn;
}

// ---------------------------------------------------------------------------

extern "C" void kernel_launch(void* const* d_in, const int* in_sizes, int n_in,
                              void* d_out, int out_size, void* d_ws, size_t ws_size,
                              hipStream_t stream) {
  const int*   tokens  = (const int*)d_in[0];
  const int*   lengths = (const int*)d_in[1];
  const float* emb     = (const float*)d_in[2];
  const float* w_ih    = (const float*)d_in[3];
  const float* w_hh    = (const float*)d_in[4];
  const float* b_hh    = (const float*)d_in[5];
  float*       out     = (float*)d_out;

  // workspace layout (all offsets multiples of 256 bytes)
  char* ws = (char*)d_ws;
  const size_t OFF_WIH = 0;                                   //  9,216,000 B
  const size_t OFF_WHH = OFF_WIH + (size_t)Gsz * EP * 2;      // 34,560,000 B
  const size_t OFF_XBF = OFF_WHH + (size_t)Gsz * Hsz * 2;     //  3,932,160 B
  const size_t OFF_GI  = OFF_XBF + (size_t)ROWS * EP * 2;     // 88,473,600 B
  const size_t OFF_GH  = OFF_GI  + (size_t)ROWS * Gsz * 4;    //  1,843,200 B
  const size_t OFF_HF  = OFF_GH  + (size_t)Bsz * Gsz * 4;     //    614,400 B
  const size_t OFF_HB  = OFF_HF  + (size_t)Bsz * Hsz * 4;     //    307,200 B
  const size_t NEED    = OFF_HB  + (size_t)Bsz * Hsz * 2;     // ~138.9 MB
  if (ws_size < NEED) return;

  bf16*  w_ih_bf = (bf16*)(ws + OFF_WIH);
  bf16*  w_hh_bf = (bf16*)(ws + OFF_WHH);
  bf16*  x_bf    = (bf16*)(ws + OFF_XBF);
  float* gi      = (float*)(ws + OFF_GI);
  float* gh      = (float*)(ws + OFF_GH);
  float* h_f     = (float*)(ws + OFF_HF);
  bf16*  h_bf    = (bf16*)(ws + OFF_HB);

  // prep
  init_h_kernel<<<(Bsz * Hsz + 255) / 256, 256, 0, stream>>>(h_f, h_bf);
  embed_kernel<<<(ROWS * EP + 255) / 256, 256, 0, stream>>>(tokens, emb, x_bf);
  conv_wih_kernel<<<(Gsz * EP + 255) / 256, 256, 0, stream>>>(w_ih, w_ih_bf);
  conv_whh_kernel<<<(Gsz * Hsz + 255) / 256, 256, 0, stream>>>(w_hh, w_hh_bf);

  // input-side projection, one big GEMM
  gi_gemm_kernel<<<dim3(Gsz / 96, ROWS / 128), 256, 0, stream>>>(x_bf, w_ih_bf, gi);

  // sequential recurrence
  for (int t = 0; t < Tsz; ++t) {
    gh_gemm_kernel<<<Gsz / 48, 128, 0, stream>>>(h_bf, w_hh_bf, gh);
    gate_kernel<<<(Bsz * Hsz + 255) / 256, 256, 0, stream>>>(gi, gh, b_hh, lengths,
                                                             h_f, h_bf, out, t);
  }
}
